// TransformerTinyRecursiveModel_41480794145077
// MI455X (gfx1250) — matrix operand order repaired and checked by reading, not compiled, not run on hardware
//
#include <hip/hip_runtime.h>
#include <hip/hip_bf16.h>

typedef __attribute__((ext_vector_type(16))) _Float16 v16h;
typedef __attribute__((ext_vector_type(8)))  _Float16 v8h;
typedef __attribute__((ext_vector_type(8)))  float    v8f;

// ---------------- workspace (f16, pre-swizzled B tiles) layout ----------------
// Each 16(N)x32(K) tile is 512 f16 stored per-lane: lane*16 + t, where
//   n = nt*16 + (lane&15),  k = kt*32 + (lane>>4)*16 + t   (same k order as load_a16)
#define OFF_WIN   0        // W_in      N=64  K=224(pad from 200) : 4*7*512 = 14336
#define OFF_ZIN   14336    // z_in_w    N=192 K=64  : 12288
#define OFF_ZOUT  26624    // z_out_w   N=64  K=64  : 4096
#define OFF_ZF1   30720    // z_ffn_w1  N=128 K=64  : 8192
#define OFF_ZF2   38912    // z_ffn_w2  N=64  K=128 : 8192
#define OFF_WVY   47104    // y_in_w[128:192] N=64 K=64 : 4096
#define OFF_YOUT  51200    // y_out_w   : 4096
#define OFF_YF1   55296    // y_ffn_w1  : 8192
#define OFF_YF2   63488    // y_ffn_w2  : 8192
#define WTS_TOTAL 71680

// LDS-resident weights = everything except W_in
#define WLDS_F16   57344          // f16 elements
#define WLDS_BYTES 114688

// per-wave scratch region (bytes), all sub-offsets 16B aligned
#define PW_Z   0        // f32 [16][68]
#define PW_Y   4352
#define PW_R   8704
#define PW_KX  13056    // f16 [16][68]
#define PW_VX  15232
#define PW_Q   17408
#define PW_KZ  19584
#define PW_VZ  21760
#define PW_KY  23936
#define PW_VY  26112
#define PW_A0  28288    // f16 [16][232]
#define PW_A1  35712    // f16 [16][136]
#define PW_SZ  40064

#define SMEM_BYTES (WLDS_BYTES + 4 * PW_SZ)   // 274944 <= 320KB/WGP

// ------------------------------- helpers -------------------------------------
union V16u { v16h v; v8h h[2]; };

__device__ __forceinline__ float gelu_f(float x) {
  return 0.5f * x * (1.0f + erff(x * 0.70710678118654752f));
}

// B tile: 32 contiguous bytes per lane (pre-swizzled)
__device__ __forceinline__ v16h load_b32B(const _Float16* w) {
  int lane = threadIdx.x & 31;
  const _Float16* p = w + lane * 16;
  V16u u;
  u.h[0] = *(const v8h*)(p);
  u.h[1] = *(const v8h*)(p + 8);
  return u.v;
}

// A tile from row-major f16 [16][sa] at k0 (sa, k0 multiples keep 16B alignment)
__device__ __forceinline__ v16h load_a16(const _Float16* A, int sa, int k0) {
  int lane = threadIdx.x & 31;
  int m  = lane & 15;
  int kh = (lane >> 4) * 16;
  const _Float16* p = A + m * sa + k0 + kh;
  V16u u;
  u.h[0] = *(const v8h*)(p);        // K = kh + 0..7
  u.h[1] = *(const v8h*)(p + 8);    // K = kh + 8..15
  return u.v;
}

template <int KT>
__device__ __forceinline__ v8f mm_tile(const _Float16* A, int sa, const _Float16* W) {
  v8f c = {0.f, 0.f, 0.f, 0.f, 0.f, 0.f, 0.f, 0.f};
#pragma unroll
  for (int kt = 0; kt < KT; ++kt) {
    v16h a = load_a16(A, sa, kt * 32);
    v16h b = load_b32B(W + kt * 512);
    c = __builtin_amdgcn_wmma_f32_16x16x32_f16(false, a, false, b, (short)0, c, false, false);
  }
  return c;
}

// C-layout element (m = r + 8*(lane>>4), n = lane&15) within a 16-col tile
__device__ __forceinline__ void store_c_f16(_Float16* dst, int sd, int col0,
                                            const float* bias, v8f c, bool dog) {
  int lane = threadIdx.x & 31;
  int n  = lane & 15;
  int mb = (lane >> 4) * 8;
  float b = bias[n];
#pragma unroll
  for (int r = 0; r < 8; ++r) {
    float v = c[r] + b;
    if (dog) v = gelu_f(v);
    dst[(mb + r) * sd + col0 + n] = (_Float16)v;
  }
}

__device__ __forceinline__ void store_c_res32(float* dst, int sd, int col0,
                                              const float* bias, const float* res, int sr,
                                              v8f c) {
  int lane = threadIdx.x & 31;
  int n  = lane & 15;
  int mb = (lane >> 4) * 8;
  float b = bias[n];
#pragma unroll
  for (int r = 0; r < 8; ++r) {
    int m = mb + r;
    float v = c[r] + b;
    if (res) v += res[m * sr + col0 + n];
    dst[m * sd + col0 + n] = v;
  }
}

__device__ __forceinline__ void f32_to_f16_64(const float* src, int ss, _Float16* dst, int sd) {
  int lane = threadIdx.x & 31;
  int row = lane & 15;
  int c0  = (lane >> 4) * 32;
#pragma unroll
  for (int i = 0; i < 32; ++i) dst[row * sd + c0 + i] = (_Float16)src[row * ss + c0 + i];
}

// LayerNorm over 64 cols of f32 [16][ss]; half-wave shuffle reduction.
__device__ __forceinline__ void layernorm16x64(const float* src, int ss,
                                               const float* g, const float* be,
                                               float* o32, int s32,
                                               _Float16* o16, int s16, bool dog) {
  int lane = threadIdx.x & 31;
  int row = lane & 15;
  int c0  = (lane >> 4) * 32;
  float v[32];
  float s = 0.f, s2 = 0.f;
#pragma unroll
  for (int i = 0; i < 32; ++i) {
    float t = src[row * ss + c0 + i];
    v[i] = t; s += t; s2 += t * t;
  }
  s  += __shfl_xor(s, 16, 32);
  s2 += __shfl_xor(s2, 16, 32);
  float mean = s * (1.f / 64.f);
  float var  = s2 * (1.f / 64.f) - mean * mean;
  float rs   = rsqrtf(var + 1e-5f);
#pragma unroll
  for (int i = 0; i < 32; ++i) {
    float t = (v[i] - mean) * rs * g[c0 + i] + be[c0 + i];
    if (dog) t = gelu_f(t);
    if (o32) o32[row * s32 + c0 + i] = t;
    if (o16) o16[row * s16 + c0 + i] = (_Float16)t;
  }
}

// --------------------------- weight prep kernel ------------------------------
__global__ void prep_w(const float* __restrict__ W, _Float16* __restrict__ dst,
                       int N, int K, int Ksrc) {
  int idx = blockIdx.x * blockDim.x + threadIdx.x;
  int tot = (N / 16) * (K / 32) * 512;
  if (idx >= tot) return;
  int tile = idx >> 9;
  int e    = idx & 511;
  int lane = e >> 4;
  int t    = e & 15;
  int ktiles = K / 32;
  int nt = tile / ktiles, kt = tile % ktiles;
  int n = nt * 16 + (lane & 15);
  int k = kt * 32 + (lane >> 4) * 16 + t;
  float v = (k < Ksrc) ? W[n * Ksrc + k] : 0.f;
  dst[idx] = (_Float16)v;
}

// ------------------------------- main kernel ---------------------------------
struct Params {
  const float *x, *b_in, *g_in, *be_in;
  const float *z_in_b, *z_out_b, *z_ffn_b1, *z_ffn_b2;
  const float *zn1_g, *zn1_b, *zn2_g, *zn2_b;
  const float *y_in_b, *y_out_b, *y_ffn_b1, *y_ffn_b2;
  const float *yn1_g, *yn1_b, *yn2_g, *yn2_b;
  const float *W_out, *b_out;
  const _Float16* wts;
  float* out;
};

__global__ void __launch_bounds__(128) trm_main(Params p) {
  extern __shared__ char smem[];
  _Float16* wsm = (_Float16*)smem;

  // Cooperative copy of swizzled f16 loop-weights (global -> LDS), 7168 uint4
  {
    const uint4* src = (const uint4*)(p.wts + OFF_ZIN);
    uint4* dst = (uint4*)wsm;
    for (int i = threadIdx.x; i < WLDS_BYTES / 16; i += blockDim.x) dst[i] = src[i];
  }
  __syncthreads();

  const _Float16* wzin  = wsm + (OFF_ZIN  - OFF_ZIN);
  const _Float16* wzout = wsm + (OFF_ZOUT - OFF_ZIN);
  const _Float16* wzf1  = wsm + (OFF_ZF1  - OFF_ZIN);
  const _Float16* wzf2  = wsm + (OFF_ZF2  - OFF_ZIN);
  const _Float16* wwvy  = wsm + (OFF_WVY  - OFF_ZIN);
  const _Float16* wyout = wsm + (OFF_YOUT - OFF_ZIN);
  const _Float16* wyf1  = wsm + (OFF_YF1  - OFF_ZIN);
  const _Float16* wyf2  = wsm + (OFF_YF2  - OFF_ZIN);

  int wave = threadIdx.x >> 5;
  int lane = threadIdx.x & 31;
  char* wr = smem + WLDS_BYTES + wave * PW_SZ;
  float*    Zf  = (float*)(wr + PW_Z);
  float*    Yf  = (float*)(wr + PW_Y);
  float*    Rf  = (float*)(wr + PW_R);
  _Float16* KXb = (_Float16*)(wr + PW_KX);
  _Float16* VXb = (_Float16*)(wr + PW_VX);
  _Float16* Qb  = (_Float16*)(wr + PW_Q);
  _Float16* KZb = (_Float16*)(wr + PW_KZ);
  _Float16* VZb = (_Float16*)(wr + PW_VZ);
  _Float16* KYb = (_Float16*)(wr + PW_KY);
  _Float16* VYb = (_Float16*)(wr + PW_VY);
  _Float16* A0  = (_Float16*)(wr + PW_A0);   // stride 232
  _Float16* A1  = (_Float16*)(wr + PW_A1);   // stride 136

  const int rowbase = (blockIdx.x * 4 + wave) * 16;
  const int row  = lane & 15;
  const int half = lane >> 4;

  // ---- load x tile (16 x 200 f32) into A0 as f16, zero-pad K to 232 ----
  {
    const float* srcp = p.x + (size_t)(rowbase + row) * 200 + half * 100;
    _Float16* dstp = A0 + row * 232 + half * 100;
#pragma unroll 4
    for (int i = 0; i < 100; ++i) dstp[i] = (_Float16)srcp[i];
    if (half) {
#pragma unroll
      for (int i = 0; i < 32; ++i) A0[row * 232 + 200 + i] = (_Float16)0.f;
    }
  }

  // ---- zero-init y, z states ----
#pragma unroll
  for (int i = 0; i < 32; ++i) {
    Zf[row * 68 + half * 32 + i] = 0.f;
    Yf[row * 68 + half * 32 + i] = 0.f;
  }

  // ---- input projection: x @ W_in^T + b_in -> LN -> gelu -> x_proj (f16 in A1)
#pragma unroll
  for (int nt = 0; nt < 4; ++nt) {
    v8f c = mm_tile<7>(A0, 232, p.wts + OFF_WIN + nt * 7 * 512);
    store_c_res32(Rf, 68, nt * 16, p.b_in + nt * 16, nullptr, 0, c);
  }
  layernorm16x64(Rf, 68, p.g_in, p.be_in, nullptr, 0, A1, 136, true);

  // ---- loop-invariant k_x, v_x (include qkv biases) ----
#pragma unroll
  for (int nt = 0; nt < 4; ++nt) {
    v8f c = mm_tile<2>(A1, 136, wzin + (4 + nt) * 1024);
    store_c_f16(KXb, 68, nt * 16, p.z_in_b + 64 + nt * 16, c, false);
  }
#pragma unroll
  for (int nt = 0; nt < 4; ++nt) {
    v8f c = mm_tile<2>(A1, 136, wzin + (8 + nt) * 1024);
    store_c_f16(VXb, 68, nt * 16, p.z_in_b + 128 + nt * 16, c, false);
  }

  // ------------------------------ 16 steps ------------------------------
#pragma unroll 1
  for (int step = 0; step < 16; ++step) {
    // qkv pass 1 (token z): q_z, k_z, v_z
    f32_to_f16_64(Zf, 68, A0, 232);
#pragma unroll
    for (int nt = 0; nt < 12; ++nt) {
      v8f c = mm_tile<2>(A0, 232, wzin + nt * 1024);
      _Float16* dst = (nt < 4) ? Qb : ((nt < 8) ? KZb : VZb);
      store_c_f16(dst, 68, (nt & 3) * 16, p.z_in_b + nt * 16, c, false);
    }
    // qkv pass 2 (token y): k_y, v_y
    f32_to_f16_64(Yf, 68, A1, 136);
#pragma unroll
    for (int nt = 4; nt < 12; ++nt) {
      v8f c = mm_tile<2>(A1, 136, wzin + nt * 1024);
      _Float16* dst = (nt < 8) ? KYb : VYb;
      store_c_f16(dst, 68, (nt & 3) * 16, p.z_in_b + nt * 16, c, false);
    }

    // attention for the z token only (3-way softmax per head) -> A0 (f16)
#pragma unroll
    for (int hh = 0; hh < 2; ++hh) {
      int h = half * 2 + hh;
      int o = h * 16;
      float q[16];
      float sx = 0.f, sy = 0.f, sz = 0.f;
#pragma unroll
      for (int d = 0; d < 16; ++d) {
        q[d] = (float)Qb[row * 68 + o + d];
        sx += q[d] * (float)KXb[row * 68 + o + d];
        sy += q[d] * (float)KYb[row * 68 + o + d];
        sz += q[d] * (float)KZb[row * 68 + o + d];
      }
      sx *= 0.25f; sy *= 0.25f; sz *= 0.25f;
      float mx = fmaxf(sx, fmaxf(sy, sz));
      float ex = __expf(sx - mx), ey = __expf(sy - mx), ez = __expf(sz - mx);
      float inv = 1.f / (ex + ey + ez);
      ex *= inv; ey *= inv; ez *= inv;
#pragma unroll
      for (int d = 0; d < 16; ++d) {
        float ov = ex * (float)VXb[row * 68 + o + d]
                 + ey * (float)VYb[row * 68 + o + d]
                 + ez * (float)VZb[row * 68 + o + d];
        A0[row * 232 + o + d] = (_Float16)ov;
      }
    }

    // z attention out-proj + residual: R = z + o @ z_out_w^T + b
#pragma unroll
    for (int nt = 0; nt < 4; ++nt) {
      v8f c = mm_tile<2>(A0, 232, wzout + nt * 1024);
      store_c_res32(Rf, 68, nt * 16, p.z_out_b + nt * 16, Zf, 68, c);
    }
    layernorm16x64(Rf, 68, p.zn1_g, p.zn1_b, Rf, 68, A1, 136, false);  // z_res

    // z FFN: gelu(z_res @ w1^T + b1) @ w2^T + b2 ; add z_res ; LN -> new z (+f16)
#pragma unroll
    for (int nt = 0; nt < 8; ++nt) {
      v8f c = mm_tile<2>(A1, 136, wzf1 + nt * 1024);
      store_c_f16(A0, 232, nt * 16, p.z_ffn_b1 + nt * 16, c, true);
    }
#pragma unroll
    for (int nt = 0; nt < 4; ++nt) {
      v8f c = mm_tile<4>(A0, 232, wzf2 + nt * 2048);
      store_c_res32(Rf, 68, nt * 16, p.z_ffn_b2 + nt * 16, Rf, 68, c);
    }
    layernorm16x64(Rf, 68, p.zn2_g, p.zn2_b, Zf, 68, A1, 136, false);  // z2

    // cross-attn path: ca = (z2 @ Wv_y^T + bv_y) @ y_out_w^T + b ; y residual
#pragma unroll
    for (int nt = 0; nt < 4; ++nt) {
      v8f c = mm_tile<2>(A1, 136, wwvy + nt * 1024);
      store_c_f16(A0, 232, nt * 16, p.y_in_b + 128 + nt * 16, c, false);
    }
#pragma unroll
    for (int nt = 0; nt < 4; ++nt) {
      v8f c = mm_tile<2>(A0, 232, wyout + nt * 1024);
      store_c_res32(Rf, 68, nt * 16, p.y_out_b + nt * 16, Yf, 68, c);
    }
    layernorm16x64(Rf, 68, p.yn1_g, p.yn1_b, Rf, 68, A1, 136, false);  // y_res

    // y FFN ; add y_res ; LN -> new y
#pragma unroll
    for (int nt = 0; nt < 8; ++nt) {
      v8f c = mm_tile<2>(A1, 136, wyf1 + nt * 1024);
      store_c_f16(A0, 232, nt * 16, p.y_ffn_b1 + nt * 16, c, true);
    }
#pragma unroll
    for (int nt = 0; nt < 4; ++nt) {
      v8f c = mm_tile<4>(A0, 232, wyf2 + nt * 2048);
      store_c_res32(Rf, 68, nt * 16, p.y_ffn_b2 + nt * 16, Rf, 68, c);
    }
    layernorm16x64(Rf, 68, p.yn2_g, p.yn2_b, Yf, 68, nullptr, 0, false);  // new y
  }

  // ---- output head: out[b] = y . W_out[0] + b_out ----
  if (lane < 16) {
    float acc = p.b_out[0];
#pragma unroll
    for (int c2 = 0; c2 < 64; ++c2) acc += Yf[lane * 68 + c2] * p.W_out[c2];
    p.out[rowbase + lane] = acc;
  }
}

// ------------------------------- launcher ------------------------------------
extern "C" void kernel_launch(void* const* d_in, const int* in_sizes, int n_in,
                              void* d_out, int out_size, void* d_ws, size_t ws_size,
                              hipStream_t stream) {
  const float* x        = (const float*)d_in[0];
  const float* W_in     = (const float*)d_in[1];
  const float* b_in     = (const float*)d_in[2];
  const float* g_in     = (const float*)d_in[3];
  const float* be_in    = (const float*)d_in[4];
  const float* z_in_w   = (const float*)d_in[5];
  const float* z_in_b   = (const float*)d_in[6];
  const float* z_out_w  = (const float*)d_in[7];
  const float* z_out_b  = (const float*)d_in[8];
  const float* z_ffn_w1 = (const float*)d_in[9];
  const float* z_ffn_b1 = (const float*)d_in[10];
  const float* z_ffn_w2 = (const float*)d_in[11];
  const float* z_ffn_b2 = (const float*)d_in[12];
  const float* zn1_g    = (const float*)d_in[13];
  const float* zn1_b    = (const float*)d_in[14];
  const float* zn2_g    = (const float*)d_in[15];
  const float* zn2_b    = (const float*)d_in[16];
  const float* y_in_w   = (const float*)d_in[17];
  const float* y_in_b   = (const float*)d_in[18];
  const float* y_out_w  = (const float*)d_in[19];
  const float* y_out_b  = (const float*)d_in[20];
  const float* y_ffn_w1 = (const float*)d_in[21];
  const float* y_ffn_b1 = (const float*)d_in[22];
  const float* y_ffn_w2 = (const float*)d_in[23];
  const float* y_ffn_b2 = (const float*)d_in[24];
  const float* yn1_g    = (const float*)d_in[25];
  const float* yn1_b    = (const float*)d_in[26];
  const float* yn2_g    = (const float*)d_in[27];
  const float* yn2_b    = (const float*)d_in[28];
  const float* W_out    = (const float*)d_in[29];
  const float* b_out    = (const float*)d_in[30];

  _Float16* wts = (_Float16*)d_ws;

  auto prep = [&](const float* W, int off, int N, int K, int Ksrc) {
    int elems = (N / 16) * (K / 32) * 512;
    prep_w<<<(elems + 255) / 256, 256, 0, stream>>>(W, wts + off, N, K, Ksrc);
  };
  prep(W_in,            OFF_WIN,  64, 224, 200);
  prep(z_in_w,          OFF_ZIN,  192, 64, 64);
  prep(z_out_w,         OFF_ZOUT, 64,  64, 64);
  prep(z_ffn_w1,        OFF_ZF1,  128, 64, 64);
  prep(z_ffn_w2,        OFF_ZF2,  64, 128, 128);
  prep(y_in_w + 128 * 64, OFF_WVY, 64, 64, 64);
  prep(y_out_w,         OFF_YOUT, 64,  64, 64);
  prep(y_ffn_w1,        OFF_YF1,  128, 64, 64);
  prep(y_ffn_w2,        OFF_YF2,  64, 128, 128);

  Params p;
  p.x = x; p.b_in = b_in; p.g_in = g_in; p.be_in = be_in;
  p.z_in_b = z_in_b; p.z_out_b = z_out_b; p.z_ffn_b1 = z_ffn_b1; p.z_ffn_b2 = z_ffn_b2;
  p.zn1_g = zn1_g; p.zn1_b = zn1_b; p.zn2_g = zn2_g; p.zn2_b = zn2_b;
  p.y_in_b = y_in_b; p.y_out_b = y_out_b; p.y_ffn_b1 = y_ffn_b1; p.y_ffn_b2 = y_ffn_b2;
  p.yn1_g = yn1_g; p.yn1_b = yn1_b; p.yn2_g = yn2_g; p.yn2_b = yn2_b;
  p.W_out = W_out; p.b_out = b_out;
  p.wts = wts; p.out = (float*)d_out;

  (void)hipFuncSetAttribute(reinterpret_cast<const void*>(&trm_main),
                            hipFuncAttributeMaxDynamicSharedMemorySize, SMEM_BYTES);
  // 131072 rows / (4 waves * 16 rows) = 2048 blocks of 128 threads (4 wave32)
  trm_main<<<2048, 128, SMEM_BYTES, stream>>>(p);
}
